// DAGT_36309653521077
// MI455X (gfx1250) — compile-verified, matrix operand order checked by reading, not run on hardware
//
#include <hip/hip_runtime.h>
#include <hip/hip_bf16.h>

// ---------------------------------------------------------------------------
// Problem constants (match reference)
// ---------------------------------------------------------------------------
#define NN   2048
#define EE   4096
#define BB   8
#define HH   512
#define LL   3
#define NHD  8
#define HD   64
#define AD   41
#define BD   10

typedef __attribute__((ext_vector_type(16))) _Float16 v16h;
typedef __attribute__((ext_vector_type(8)))  _Float16 v8h;
typedef __attribute__((ext_vector_type(4)))  _Float16 v4h;
typedef __attribute__((ext_vector_type(8)))  float    v8f;
typedef __attribute__((ext_vector_type(4)))  float    v4f;

// ---------------------------------------------------------------------------
// Helpers
// ---------------------------------------------------------------------------
__device__ inline v8f zero8() {
  v8f z;
#pragma unroll
  for (int i = 0; i < 8; ++i) z[i] = 0.0f;
  return z;
}

__device__ inline float gelu_f(float x) {
  return 0.5f * x * (1.0f + erff(x * 0.70710678118654752f));
}

__device__ inline v8f wmma_f16(v16h a, v16h b, v8f c) {
  // D = A(16x32 f16) * B(32x16 f16) + C(16x16 f32)
  return __builtin_amdgcn_wmma_f32_16x16x32_f16(
      /*neg_a=*/false, a, /*neg_b=*/false, b,
      /*c_mod=*/(short)0, c, /*reuse_a=*/false, /*reuse_b=*/false);
}

// CDNA5 16-bit A-matrix fragment layout, 16x32 (rows x K):
// lane L -> row = L%16, half = L/16;
// elements 0..7  hold k = half*8 + 0..7        (16 contiguous bytes)
// elements 8..15 hold k = 16 + half*8 + 0..7   (16 contiguous bytes)
// => two ds_load_b128 per fragment when K is contiguous in LDS.
__device__ inline v16h lds_frag_row(const _Float16* __restrict__ p, int rstride,
                                    int lane) {
  const int r = lane & 15;
  const int half = lane >> 4;
  const _Float16* rp = p + r * rstride + half * 8;
  v8h lo = *(const v8h*)(rp);       // k = half*8 .. +7
  v8h hi = *(const v8h*)(rp + 16);  // k = 16+half*8 .. +7
  v16h f;
#pragma unroll
  for (int i = 0; i < 8; ++i) {
    f[i]     = lo[i];
    f[8 + i] = hi[i];
  }
  return f;
}

// Generic (strided-K) fragment gather; used for V^T where K strides LDS rows.
__device__ inline v16h lds_frag(const _Float16* __restrict__ p, int rstride,
                                int kstride, int lane) {
  const int r = lane & 15;
  const int half = lane >> 4;
  const _Float16* rp = p + r * rstride;
  v16h f;
#pragma unroll
  for (int v = 0; v < 8; ++v) {
    int k0 = (v < 4) ? (half * 8 + 2 * v) : (16 + half * 8 + 2 * (v - 4));
    f[2 * v]     = rp[k0 * kstride];
    f[2 * v + 1] = rp[(k0 + 1) * kstride];
  }
  return f;
}

// A-layout fragment from global f32 with scale.
__device__ inline v16h gfrag_f32(const float* __restrict__ p, int ld, int lane,
                                 float scale) {
  const int r = lane & 15;
  const int half = lane >> 4;
  const float* rp = p + (size_t)r * ld;
  v16h f;
#pragma unroll
  for (int v = 0; v < 8; ++v) {
    int k0 = (v < 4) ? (half * 8 + 2 * v) : (16 + half * 8 + 2 * (v - 4));
    f[2 * v]     = (_Float16)(rp[k0] * scale);
    f[2 * v + 1] = (_Float16)(rp[k0 + 1] * scale);
  }
  return f;
}

__device__ inline v4h cvt4(v4f a) {
  v4h h;
  h[0] = (_Float16)a[0];
  h[1] = (_Float16)a[1];
  h[2] = (_Float16)a[2];
  h[3] = (_Float16)a[3];
  return h;
}

// ---------------------------------------------------------------------------
// Generic WMMA GEMM: C[M,N] = act(A[M,K] @ B[N,K]^T + bias[N])
// Block: 256 threads = 8 waves; block tile 128(M) x 64(N);
// wave grid 4x2, wave tile 32x32 -> 4 WMMAs per 32-wide k-step.
// Fast path is software-pipelined: tile kk+32 is prefetched into registers
// while tile kk is consumed by the WMMAs. Slow path uses clamped
// (data-predicated) loads: addresses stay in range, OOB values select to 0.
// ---------------------------------------------------------------------------
#define OP_NONE 0
#define OP_GELU 1

__global__ __launch_bounds__(256)
void gemm_wmma_kernel(const float* __restrict__ A, int lda,
                      const float* __restrict__ B, int ldb,
                      const float* __restrict__ bias,
                      float* __restrict__ C, int ldc,
                      int M, int N, int K, int op) {
  __shared__ _Float16 As[128][40];
  __shared__ _Float16 Bs[64][40];

  const int tid  = threadIdx.x;
  const int lane = tid & 31;
  const int wave = tid >> 5;   // 0..7
  const int wm   = wave >> 1;  // 0..3 -> M offset wm*32
  const int wn   = wave & 1;   // 0..1 -> N offset wn*32
  const int m0   = blockIdx.y * 128;
  const int n0   = blockIdx.x * 64;

  v8f acc00 = zero8(), acc01 = zero8();  // rows wm*32..+16
  v8f acc10 = zero8(), acc11 = zero8();  // rows wm*32+16..+32

  // Per-thread staging coordinates (A: 128x32 tile, B: 64x32 tile).
  const int ar0 = tid >> 3;        // A rows: ar0, ar0+32, ar0+64, ar0+96
  const int ac4 = (tid & 7) * 4;   // A col group
  const bool fast_path = (m0 + 128 <= M) && (n0 + 64 <= N) &&
                         ((lda & 3) == 0) && ((ldb & 3) == 0) &&
                         ((K & 31) == 0);

  if (fast_path) {
    v4f at[4], bt[2];
    // Prime the pipeline: tile kk = 0.
#pragma unroll
    for (int i = 0; i < 4; ++i)
      at[i] = *(const v4f*)(A + (size_t)(m0 + ar0 + 32 * i) * lda + ac4);
#pragma unroll
    for (int i = 0; i < 2; ++i)
      bt[i] = *(const v4f*)(B + (size_t)(n0 + ar0 + 32 * i) * ldb + ac4);

    for (int kk = 0; kk < K; kk += 32) {
      // Commit prefetched tile to LDS.
#pragma unroll
      for (int i = 0; i < 4; ++i)
        *(v4h*)&As[ar0 + 32 * i][ac4] = cvt4(at[i]);
#pragma unroll
      for (int i = 0; i < 2; ++i)
        *(v4h*)&Bs[ar0 + 32 * i][ac4] = cvt4(bt[i]);
      __syncthreads();

      // Prefetch next tile (overlaps the WMMAs below).
      if (kk + 32 < K) {
        int kn = kk + 32;
#pragma unroll
        for (int i = 0; i < 4; ++i)
          at[i] =
              *(const v4f*)(A + (size_t)(m0 + ar0 + 32 * i) * lda + kn + ac4);
#pragma unroll
        for (int i = 0; i < 2; ++i)
          bt[i] =
              *(const v4f*)(B + (size_t)(n0 + ar0 + 32 * i) * ldb + kn + ac4);
      }

      v16h af0 = lds_frag_row(&As[wm * 32][0], 40, lane);
      v16h af1 = lds_frag_row(&As[wm * 32 + 16][0], 40, lane);
      v16h bf0 = lds_frag_row(&Bs[wn * 32][0], 40, lane);
      v16h bf1 = lds_frag_row(&Bs[wn * 32 + 16][0], 40, lane);
      acc00 = wmma_f16(af0, bf0, acc00);
      acc01 = wmma_f16(af0, bf1, acc01);
      acc10 = wmma_f16(af1, bf0, acc10);
      acc11 = wmma_f16(af1, bf1, acc11);
      __syncthreads();
    }
  } else {
    // Data-predicated path: clamp indices (always-legal addresses), then
    // select OOB lanes to zero. Plain batched loads, no exec-mask dances.
    const int mclamp = M - 1;
    const int nclamp = N - 1;
    const int kclamp = K - 1;
    for (int kk = 0; kk < K; kk += 32) {
      float av[16], bv[8];
#pragma unroll
      for (int i = 0; i < 16; ++i) {
        int l  = tid + 256 * i;  // 0..4095
        int rr = l >> 5;         // 0..127
        int cc = l & 31;
        int gm = m0 + rr, gk = kk + cc;
        int cm = gm < mclamp ? gm : mclamp;
        int ck = gk < kclamp ? gk : kclamp;
        av[i] = A[(size_t)cm * lda + ck];
        if (gm >= M || gk >= K) av[i] = 0.0f;
      }
#pragma unroll
      for (int i = 0; i < 8; ++i) {
        int l  = tid + 256 * i;  // 0..2047
        int rr = l >> 5;         // 0..63
        int cc = l & 31;
        int gn = n0 + rr, gk = kk + cc;
        int cn2 = gn < nclamp ? gn : nclamp;
        int ck  = gk < kclamp ? gk : kclamp;
        bv[i] = B[(size_t)cn2 * ldb + ck];
        if (gn >= N || gk >= K) bv[i] = 0.0f;
      }
#pragma unroll
      for (int i = 0; i < 16; ++i) {
        int l = tid + 256 * i;
        As[l >> 5][l & 31] = (_Float16)av[i];
      }
#pragma unroll
      for (int i = 0; i < 8; ++i) {
        int l = tid + 256 * i;
        Bs[l >> 5][l & 31] = (_Float16)bv[i];
      }
      __syncthreads();
      v16h af0 = lds_frag_row(&As[wm * 32][0], 40, lane);
      v16h af1 = lds_frag_row(&As[wm * 32 + 16][0], 40, lane);
      v16h bf0 = lds_frag_row(&Bs[wn * 32][0], 40, lane);
      v16h bf1 = lds_frag_row(&Bs[wn * 32 + 16][0], 40, lane);
      acc00 = wmma_f16(af0, bf0, acc00);
      acc01 = wmma_f16(af0, bf1, acc01);
      acc10 = wmma_f16(af1, bf0, acc10);
      acc11 = wmma_f16(af1, bf1, acc11);
      __syncthreads();
    }
  }

  // Epilogue: C layout -> lane L: n = L%16, rows = (L/16)*8 + vgpr.
  const int cn = lane & 15;
  const int mh = (lane >> 4) * 8;
  const int na = n0 + wn * 32 + cn;
  const int nb = na + 16;
  const float ba = (na < N) ? bias[na] : 0.0f;
  const float bb = (nb < N) ? bias[nb] : 0.0f;
#pragma unroll
  for (int sub = 0; sub < 2; ++sub) {
    const v8f& a0 = sub ? acc10 : acc00;
    const v8f& a1 = sub ? acc11 : acc01;
#pragma unroll
    for (int r = 0; r < 8; ++r) {
      int m = m0 + wm * 32 + sub * 16 + mh + r;
      if (m >= M) continue;
      if (na < N) {
        float v = a0[r] + ba;
        if (op == OP_GELU) v = gelu_f(v);
        C[(size_t)m * ldc + na] = v;
      }
      if (nb < N) {
        float v = a1[r] + bb;
        if (op == OP_GELU) v = gelu_f(v);
        C[(size_t)m * ldc + nb] = v;
      }
    }
  }
}

// ---------------------------------------------------------------------------
// Flash attention (one block = one 64-row q tile of one head; 4 waves).
// qkv: [seq, 3H] (q|k|v), out: [seq, H]; hd = 64, online softmax,
// K/V tile prefetch pipelined against score/softmax/PV compute.
// ---------------------------------------------------------------------------
__global__ __launch_bounds__(128)
void flash_attn_kernel(const float* __restrict__ qkv, int ld,
                       float* __restrict__ out, int ldo, int seq) {
  __shared__ _Float16 Ks[32][72];
  __shared__ _Float16 Vs[32][72];
  __shared__ _Float16 Ps[4][16][32];

  const int tid  = threadIdx.x;
  const int lane = tid & 31;
  const int wave = tid >> 5;  // 0..3
  const int head = blockIdx.y;
  const int q0   = blockIdx.x * 64;
  const int qrow0 = q0 + wave * 16;

  const float* Qb = qkv + (size_t)qrow0 * ld + head * HD;
  const float* Kb = qkv + head * HD + HH;
  const float* Vb = qkv + head * HD + 2 * HH;

  const float scale = 0.125f;  // 1/sqrt(64)
  const v16h qf0 = gfrag_f32(Qb, ld, lane, scale);       // hd 0..31
  const v16h qf1 = gfrag_f32(Qb + 32, ld, lane, scale);  // hd 32..63

  float mrow[8], lrow[8];
#pragma unroll
  for (int i = 0; i < 8; ++i) { mrow[i] = -1.0e30f; lrow[i] = 0.0f; }
  v8f o0 = zero8(), o1 = zero8(), o2 = zero8(), o3 = zero8();

  const int mh = (lane >> 4) * 8;
  const int kr = tid >> 4;        // staging row 0..7 (+8*i)
  const int kc4 = (tid & 15) * 4; // staging col group

  // Prime K/V prefetch for tile 0: 32x64 = 512 float4 each, 4 per thread.
  v4f kt[4], vt[4];
#pragma unroll
  for (int i = 0; i < 4; ++i) {
    kt[i] = *(const v4f*)(Kb + (size_t)(kr + 8 * i) * ld + kc4);
    vt[i] = *(const v4f*)(Vb + (size_t)(kr + 8 * i) * ld + kc4);
  }

  for (int k0 = 0; k0 < seq; k0 += 32) {
    // Commit prefetched K/V to LDS.
#pragma unroll
    for (int i = 0; i < 4; ++i) {
      *(v4h*)&Ks[kr + 8 * i][kc4] = cvt4(kt[i]);
      *(v4h*)&Vs[kr + 8 * i][kc4] = cvt4(vt[i]);
    }
    __syncthreads();

    // Prefetch next K/V tile (overlaps everything below).
    if (k0 + 32 < seq) {
      int kn = k0 + 32;
#pragma unroll
      for (int i = 0; i < 4; ++i) {
        kt[i] = *(const v4f*)(Kb + (size_t)(kn + kr + 8 * i) * ld + kc4);
        vt[i] = *(const v4f*)(Vb + (size_t)(kn + kr + 8 * i) * ld + kc4);
      }
    }

    // Scores: S[16q x 32k] = Q (16x64) * K^T, via 2 k-steps of 32.
    v8f s0 = zero8(), s1 = zero8();
    {
      v16h b0 = lds_frag_row(&Ks[0][0], 72, lane);
      v16h b1 = lds_frag_row(&Ks[0][32], 72, lane);
      s0 = wmma_f16(qf0, b0, s0);
      s0 = wmma_f16(qf1, b1, s0);
      v16h b2 = lds_frag_row(&Ks[16][0], 72, lane);
      v16h b3 = lds_frag_row(&Ks[16][32], 72, lane);
      s1 = wmma_f16(qf0, b2, s1);
      s1 = wmma_f16(qf1, b3, s1);
    }

    // Online softmax per q row (row = spread over a 16-lane half).
#pragma unroll
    for (int r = 0; r < 8; ++r) {
      float x = fmaxf(s0[r], s1[r]);
      x = fmaxf(x, __shfl_xor(x, 1));
      x = fmaxf(x, __shfl_xor(x, 2));
      x = fmaxf(x, __shfl_xor(x, 4));
      x = fmaxf(x, __shfl_xor(x, 8));
      float mn   = fmaxf(mrow[r], x);
      float corr = __expf(mrow[r] - mn);
      mrow[r] = mn;
      float p0 = __expf(s0[r] - mn);
      float p1 = __expf(s1[r] - mn);
      float rs = p0 + p1;
      rs += __shfl_xor(rs, 1);
      rs += __shfl_xor(rs, 2);
      rs += __shfl_xor(rs, 4);
      rs += __shfl_xor(rs, 8);
      lrow[r] = lrow[r] * corr + rs;
      o0[r] = o0[r] * corr;
      o1[r] = o1[r] * corr;
      o2[r] = o2[r] * corr;
      o3[r] = o3[r] * corr;
      // Stash P (C-layout -> LDS) for A-fragment reload.
      int rowi = mh + r;
      Ps[wave][rowi][lane & 15]        = (_Float16)p0;
      Ps[wave][rowi][16 + (lane & 15)] = (_Float16)p1;
    }
    __syncthreads();

    // O += P (16x32) * V (32x64): 4 WMMAs along hd.
    v16h pf  = lds_frag_row(&Ps[wave][0][0], 32, lane);
    v16h vf0 = lds_frag(&Vs[0][0],  1, 72, lane);  // B^T rows = hd, K = key
    v16h vf1 = lds_frag(&Vs[0][16], 1, 72, lane);
    v16h vf2 = lds_frag(&Vs[0][32], 1, 72, lane);
    v16h vf3 = lds_frag(&Vs[0][48], 1, 72, lane);
    o0 = wmma_f16(pf, vf0, o0);
    o1 = wmma_f16(pf, vf1, o1);
    o2 = wmma_f16(pf, vf2, o2);
    o3 = wmma_f16(pf, vf3, o3);
    __syncthreads();
  }

  const int cn = lane & 15;
#pragma unroll
  for (int r = 0; r < 8; ++r) {
    float inv = 1.0f / lrow[r];
    int m = qrow0 + mh + r;
    float* op = out + (size_t)m * ldo + head * HD;
    op[cn]      = o0[r] * inv;
    op[16 + cn] = o1[r] * inv;
    op[32 + cn] = o2[r] * inv;
    op[48 + cn] = o3[r] * inv;
  }
}

// ---------------------------------------------------------------------------
// Glue kernels
// ---------------------------------------------------------------------------
__global__ void deg_kernel(const int* __restrict__ dst, float* __restrict__ deg,
                           int E) {
  int e = blockIdx.x * blockDim.x + threadIdx.x;
  if (e < E) atomicAdd(&deg[dst[e]], 1.0f);
}

__global__ void scatter_add_kernel(const float* __restrict__ h,
                                   const int* __restrict__ dst,
                                   float* __restrict__ S, int ldS, int coff,
                                   int E, int H) {
  int i = blockIdx.x * blockDim.x + threadIdx.x;
  if (i >= E * H) return;
  int e = i / H;
  int k = i - e * H;
  atomicAdd(&S[(size_t)dst[e] * ldS + coff + k], h[i]);
}

__global__ void r_kernel(const float* __restrict__ S,
                         const float* __restrict__ deg,
                         const int* __restrict__ dst,
                         const float* __restrict__ h, float* __restrict__ r,
                         int E, int H) {
  int i = blockIdx.x * blockDim.x + threadIdx.x;
  if (i >= E * H) return;
  int e = i / H;
  int k = i - e * H;
  int d = dst[e];
  r[i] = S[(size_t)d * H + k] - deg[d] * h[i];
}

// y = a + 2*r (in place on a is fine)
__global__ void add2_kernel(float* __restrict__ a, const float* __restrict__ r,
                            int n) {
  int i = blockIdx.x * blockDim.x + threadIdx.x;
  if (i < n) a[i] = a[i] + 2.0f * r[i];
}

__global__ __launch_bounds__(256)
void layernorm_kernel(const float* __restrict__ x, int ldx,
                      const float* __restrict__ g, const float* __restrict__ b,
                      float* __restrict__ y, int ldy, int H, int op) {
  __shared__ float sA[256];
  __shared__ float sB[256];
  const int row = blockIdx.x;
  const float* xr = x + (size_t)row * ldx;
  float s = 0.0f, s2 = 0.0f;
  for (int k = threadIdx.x; k < H; k += blockDim.x) {
    float v = xr[k];
    s += v;
    s2 += v * v;
  }
  sA[threadIdx.x] = s;
  sB[threadIdx.x] = s2;
  __syncthreads();
  for (int off = 128; off > 0; off >>= 1) {
    if (threadIdx.x < off) {
      sA[threadIdx.x] += sA[threadIdx.x + off];
      sB[threadIdx.x] += sB[threadIdx.x + off];
    }
    __syncthreads();
  }
  float mean = sA[0] / (float)H;
  float var  = sB[0] / (float)H - mean * mean;
  float inv  = rsqrtf(var + 1e-5f);
  for (int k = threadIdx.x; k < H; k += blockDim.x) {
    float v = (xr[k] - mean) * inv * g[k] + b[k];
    if (op == OP_GELU) v = gelu_f(v);
    y[(size_t)row * ldy + k] = v;
  }
}

__global__ void pool_cnt_kernel(const int* __restrict__ batch,
                                float* __restrict__ cnt, int N) {
  int i = blockIdx.x * blockDim.x + threadIdx.x;
  if (i < N) atomicAdd(&cnt[batch[i]], 1.0f);
}

__global__ void pool_accum_kernel(const float* __restrict__ hn,
                                  const int* __restrict__ batch,
                                  float* __restrict__ pool, int N, int H) {
  int i = blockIdx.x * blockDim.x + threadIdx.x;
  if (i >= N * H) return;
  int n = i / H;
  int k = i - n * H;
  atomicAdd(&pool[(size_t)batch[n] * H + k], hn[i]);
}

__global__ void pool_div_kernel(const float* __restrict__ pool,
                                const float* __restrict__ cnt,
                                float* __restrict__ hg, int B, int H) {
  int i = blockIdx.x * blockDim.x + threadIdx.x;
  if (i >= B * H) return;
  hg[i] = pool[i] / cnt[i / H];
}

// ---------------------------------------------------------------------------
// Host-side launcher
// ---------------------------------------------------------------------------
static inline void gemm(const float* A, int lda, const float* B, int ldb,
                        const float* bias, float* C, int ldc, int M, int N,
                        int K, int op, hipStream_t s) {
  dim3 grid((N + 63) / 64, (M + 127) / 128);
  gemm_wmma_kernel<<<grid, 256, 0, s>>>(A, lda, B, ldb, bias, C, ldc, M, N, K,
                                        op);
}

extern "C" void kernel_launch(void* const* d_in, const int* in_sizes, int n_in,
                              void* d_out, int out_size, void* d_ws,
                              size_t ws_size, hipStream_t stream) {
  (void)in_sizes; (void)n_in; (void)out_size; (void)ws_size;

  const float* x          = (const float*)d_in[0];
  const int*   edge_index = (const int*)d_in[1];
  const float* edge_attr  = (const float*)d_in[2];
  const int*   batch      = (const int*)d_in[3];
  const float* bond_W     = (const float*)d_in[4];
  const float* bond_b     = (const float*)d_in[5];
  const float* Wh_W       = (const float*)d_in[6];
  const float* Wh_b       = (const float*)d_in[7];
  const float* attn_ln_g  = (const float*)d_in[8];
  const float* attn_ln_b  = (const float*)d_in[9];
  const float* qkv_W      = (const float*)d_in[10];
  const float* qkv_b      = (const float*)d_in[11];
  const float* attn_o_W   = (const float*)d_in[12];
  const float* attn_o_b   = (const float*)d_in[13];
  const float* upd_W      = (const float*)d_in[14];
  const float* upd_b      = (const float*)d_in[15];
  const float* upd_ln_g   = (const float*)d_in[16];
  const float* upd_ln_b   = (const float*)d_in[17];
  const float* atom_W     = (const float*)d_in[18];
  const float* atom_b     = (const float*)d_in[19];
  const float* atom_ln_g  = (const float*)d_in[20];
  const float* atom_ln_b  = (const float*)d_in[21];
  const float* feat_W     = (const float*)d_in[22];
  const float* feat_b     = (const float*)d_in[23];
  const float* a_ln_g     = (const float*)d_in[24];
  const float* a_ln_b     = (const float*)d_in[25];
  const float* a_qkv_W    = (const float*)d_in[26];
  const float* a_qkv_b    = (const float*)d_in[27];
  const float* a_out_W    = (const float*)d_in[28];
  const float* a_out_b    = (const float*)d_in[29];
  const float* gp1_W      = (const float*)d_in[30];
  const float* gp1_b      = (const float*)d_in[31];
  const float* gp_ln_g    = (const float*)d_in[32];
  const float* gp_ln_b    = (const float*)d_in[33];
  const float* gp2_W      = (const float*)d_in[34];
  const float* gp2_b      = (const float*)d_in[35];

  const int* dst = edge_index + EE;  // edge_index row 1

  // Workspace carve-up (floats).
  float* ws   = (float*)d_ws;
  float* h    = ws;                  // [E,H]
  float* S    = h    + (size_t)EE * HH;      // [N,H]
  float* deg  = S    + (size_t)NN * HH;      // [N]
  float* r    = deg  + NN;                   // [E,H]
  float* rln  = r    + (size_t)EE * HH;      // [E,H]  (also GEMM temp / u)
  float* qkvb = rln  + (size_t)EE * HH;      // [E,3H]
  float* aout = qkvb + (size_t)EE * 3 * HH;  // [E,H]  (attn ctx / atom emb tmp)
  float* abuf = aout + (size_t)EE * HH;      // [E,H]  (attn out proj / tij)
  float* cat  = abuf + (size_t)EE * HH;      // [N,2H] (concat a_i | h_i_bond)
  float* xi   = cat  + (size_t)NN * 2 * HH;  // [N,H]
  float* pool = xi   + (size_t)NN * HH;      // [B,H]
  float* cnt  = pool + (size_t)BB * HH;      // [B]
  float* hg   = cnt  + BB;                   // [B,H]
  float* g1   = hg   + (size_t)BB * HH;      // [B,H]

  const int EH  = EE * HH;
  const int NH_ = NN * HH;
  dim3 blk(256);
  dim3 grid_eh((EH + 255) / 256);
  dim3 grid_nh((NH_ + 255) / 256);

  // ---- bond embedding: h = gelu(edge_attr @ bondW^T + b) @ WhW^T + Whb
  gemm(edge_attr, BD, bond_W, BD, bond_b, rln, HH, EE, HH, BD, OP_GELU, stream);
  gemm(rln, HH, Wh_W, HH, Wh_b, h, HH, EE, HH, HH, OP_NONE, stream);

  // ---- degree
  hipMemsetAsync(deg, 0, NN * sizeof(float), stream);
  deg_kernel<<<(EE + 255) / 256, blk, 0, stream>>>(dst, deg, EE);

  // ---- message passing layers
  for (int t = 0; t < LL; ++t) {
    hipMemsetAsync(S, 0, (size_t)NN * HH * sizeof(float), stream);
    scatter_add_kernel<<<grid_eh, blk, 0, stream>>>(h, dst, S, HH, 0, EE, HH);
    r_kernel<<<grid_eh, blk, 0, stream>>>(S, deg, dst, h, r, EE, HH);

    layernorm_kernel<<<EE, blk, 0, stream>>>(r, HH, attn_ln_g + t * HH,
                                             attn_ln_b + t * HH, rln, HH, HH,
                                             OP_NONE);
    gemm(rln, HH, qkv_W + (size_t)t * 3 * HH * HH, HH, qkv_b + t * 3 * HH,
         qkvb, 3 * HH, EE, 3 * HH, HH, OP_NONE, stream);

    dim3 fgrid(EE / 64, NHD);
    flash_attn_kernel<<<fgrid, 128, 0, stream>>>(qkvb, 3 * HH, aout, HH, EE);

    gemm(aout, HH, attn_o_W + (size_t)t * HH * HH, HH, attn_o_b + t * HH, abuf,
         HH, EE, HH, HH, OP_NONE, stream);
    add2_kernel<<<grid_eh, blk, 0, stream>>>(abuf, r, EH);  // tij = a + 2r
    gemm(abuf, HH, upd_W + (size_t)t * HH * HH, HH, upd_b + t * HH, rln, HH,
         EE, HH, HH, OP_NONE, stream);
    layernorm_kernel<<<EE, blk, 0, stream>>>(rln, HH, upd_ln_g + t * HH,
                                             upd_ln_b + t * HH, h, HH, HH,
                                             OP_GELU);
  }

  // ---- atom stage
  hipMemsetAsync(cat, 0, (size_t)NN * 2 * HH * sizeof(float), stream);
  scatter_add_kernel<<<grid_eh, blk, 0, stream>>>(h, dst, cat, 2 * HH, HH, EE,
                                                  HH);  // h_i_bond -> cols H..
  gemm(x, AD, atom_W, AD, atom_b, aout, HH, NN, HH, AD, OP_NONE, stream);
  layernorm_kernel<<<NN, blk, 0, stream>>>(aout, HH, atom_ln_g, atom_ln_b, cat,
                                           2 * HH, HH, OP_GELU);  // a_i
  gemm(cat, 2 * HH, feat_W, 2 * HH, feat_b, xi, HH, NN, HH, 2 * HH, OP_NONE,
       stream);

  layernorm_kernel<<<NN, blk, 0, stream>>>(xi, HH, a_ln_g, a_ln_b, r, HH, HH,
                                           OP_NONE);
  gemm(r, HH, a_qkv_W, HH, a_qkv_b, qkvb, 3 * HH, NN, 3 * HH, HH, OP_NONE,
       stream);
  dim3 agrid(NN / 64, NHD);
  flash_attn_kernel<<<agrid, 128, 0, stream>>>(qkvb, 3 * HH, aout, HH, NN);
  gemm(aout, HH, a_out_W, HH, a_out_b, abuf, HH, NN, HH, HH, OP_NONE, stream);
  add2_kernel<<<grid_nh, blk, 0, stream>>>(abuf, xi, NH_);  // h_node

  // ---- per-graph mean pooling
  hipMemsetAsync(pool, 0, (size_t)BB * HH * sizeof(float), stream);
  hipMemsetAsync(cnt, 0, BB * sizeof(float), stream);
  pool_cnt_kernel<<<(NN + 255) / 256, blk, 0, stream>>>(batch, cnt, NN);
  pool_accum_kernel<<<grid_nh, blk, 0, stream>>>(abuf, batch, pool, NN, HH);
  pool_div_kernel<<<(BB * HH + 255) / 256, blk, 0, stream>>>(pool, cnt, hg, BB,
                                                             HH);

  // ---- head MLP
  gemm(hg, HH, gp1_W, HH, gp1_b, g1, HH, BB, HH, HH, OP_NONE, stream);
  layernorm_kernel<<<BB, blk, 0, stream>>>(g1, HH, gp_ln_g, gp_ln_b, hg, HH,
                                           HH, OP_GELU);
  gemm(hg, HH, gp2_W, HH, gp2_b, (float*)d_out, HH, BB, HH, HH, OP_NONE,
       stream);
}
